// MultiPositiveInfoNCELoss_48387101556850
// MI455X (gfx1250) — matrix operand order repaired and checked by reading, not compiled, not run on hardware
//
#include <hip/hip_runtime.h>

#define S_DIM 4
#define N_DIM 4096
#define M_DIM 4096
#define D_DIM 128
#define NEG_INF (-__builtin_inff())

typedef __attribute__((ext_vector_type(16))) _Float16 v16h;
typedef __attribute__((ext_vector_type(8)))  _Float16 v8h;
typedef __attribute__((ext_vector_type(4)))  _Float16 v4h;
typedef __attribute__((ext_vector_type(8)))  float    v8f;
typedef __attribute__((ext_vector_type(4)))  float    v4f;
typedef __attribute__((ext_vector_type(4)))  int      v4i;

#if __has_builtin(__builtin_amdgcn_global_load_async_to_lds_b128)
#define HAVE_ASYNC_LDS 1
#else
#define HAVE_ASYNC_LDS 0
#endif

__device__ __forceinline__ void wait_async_lds() {
#if HAVE_ASYNC_LDS
#if __has_builtin(__builtin_amdgcn_s_wait_asynccnt)
    __builtin_amdgcn_s_wait_asynccnt(0);
#else
    asm volatile("s_wait_asynccnt 0x0" ::: "memory");
#endif
#endif
}

// ---------------------------------------------------------------------------
// Kernel 1: L2-normalize rows of art/ref and convert to f16.
// One wave (32 lanes) per row of 128 floats; 4 floats per lane.
// ---------------------------------------------------------------------------
__global__ __launch_bounds__(256) void nrm_kernel(
    const float* __restrict__ art, const float* __restrict__ ref,
    _Float16* __restrict__ aH, _Float16* __restrict__ rH)
{
    const int lane = threadIdx.x & 31;
    const int wave = threadIdx.x >> 5;
    const long long row  = (long long)blockIdx.x * 8 + wave;
    const long long nA   = (long long)S_DIM * N_DIM;
    const long long nTot = nA + (long long)S_DIM * M_DIM;
    if (row >= nTot) return;

    const float* src;
    _Float16*    dst;
    if (row < nA) { src = art + row * D_DIM;        dst = aH + row * D_DIM; }
    else          { src = ref + (row - nA) * D_DIM; dst = rH + (row - nA) * D_DIM; }

    v4f x = *(const v4f*)(src + lane * 4);
    float ss = x[0]*x[0] + x[1]*x[1] + x[2]*x[2] + x[3]*x[3];
#pragma unroll
    for (int off = 16; off > 0; off >>= 1) ss += __shfl_xor(ss, off, 32);
    const float scale = 1.0f / fmaxf(sqrtf(ss), 1e-12f);

    v4h h;
    h[0] = (_Float16)(x[0] * scale);
    h[1] = (_Float16)(x[1] * scale);
    h[2] = (_Float16)(x[2] * scale);
    h[3] = (_Float16)(x[3] * scale);
    *(v4h*)(dst + lane * 4) = h;
}

// ---------------------------------------------------------------------------
// Kernel 2: fused cosine-GEMM (WMMA f16->f32) + streaming logsumexp.
// 8 waves/block, 16 anchors/wave (WMMA B/N dim). Ref rows stream through a
// double-buffered LDS tile (async global->LDS when available) as the WMMA
// A/M dim; each C lane holds logits of one anchor -> per-lane online softmax.
// ---------------------------------------------------------------------------
__global__ __launch_bounds__(256) void infonce_kernel(
    const _Float16* __restrict__ aH, const _Float16* __restrict__ rH,
    const unsigned char* __restrict__ pm, const float* __restrict__ logT,
    float* __restrict__ stemSum, int* __restrict__ stemCnt)
{
    __shared__ _Float16 lds[2][128 * 136];   // 2 x (128 ref rows x 128 K, +8 pad)

    const int s    = blockIdx.y;
    const int lane = threadIdx.x & 31;
    const int wave = threadIdx.x >> 5;
    const int n0     = blockIdx.x * 128 + wave * 16;
    const int anchor = n0 + (lane & 15);
    const float invT = __expf(-logT[0]);   // 1/temp

    const _Float16* rstem = rH + (long long)s * M_DIM * D_DIM;

    // Stage one 128-row ref tile into LDS buffer `buf` (16B per lane per iter).
    auto stage = [&](int buf, int m0) {
        const _Float16* rbase = rstem + (long long)m0 * D_DIM;
#pragma unroll
        for (int i = 0; i < 8; ++i) {
            int idx = i * 256 + threadIdx.x;     // 0..2047
            int rr  = idx >> 4;
            int kk  = (idx & 15) * 8;
#if HAVE_ASYNC_LDS
            __builtin_amdgcn_global_load_async_to_lds_b128(
                (__attribute__((address_space(1))) v4i*)(rbase + rr * D_DIM + kk),
                (__attribute__((address_space(3))) v4i*)(&lds[buf][rr * 136 + kk]),
                0, 0);
#else
            *(v8h*)(&lds[buf][rr * 136 + kk]) = *(const v8h*)(rbase + rr * D_DIM + kk);
#endif
        }
    };

    // Resident anchor tiles (B operand). k order used here (contiguous, same as A):
    // lane n (0-15): col n, K=kc*32+0..15 ; lane n+16: col n, K=kc*32+16..31.
    v16h breg[4];
    {
        const _Float16* abase =
            aH + ((long long)s * N_DIM + anchor) * D_DIM + (lane >> 4) * 16;
#pragma unroll
        for (int kc = 0; kc < 4; ++kc)
            breg[kc] = *(const v16h*)(abase + kc * 32);
    }

    float mden = NEG_INF, sden = 0.f;   // running logsumexp over all M
    float mnum = NEG_INF, snum = 0.f;   // running logsumexp over positives
    int   npos = 0;

    const unsigned long long* mrow = (const unsigned long long*)
        (pm + ((long long)s * N_DIM + anchor) * (long long)M_DIM);
    const int mqoff = (lane < 16) ? 0 : 1;   // which 8-byte half of a 16-col subtile

    constexpr int NT = M_DIM / 128;   // 32 tiles
    stage(0, 0);

    for (int t = 0; t < NT; ++t) {
        wait_async_lds();
        __syncthreads();                       // tile t resident; prev reads done
        if (t + 1 < NT) stage((t + 1) & 1, (t + 1) * 128);

        const _Float16* lbase = &lds[t & 1][0];
        const int m0 = t * 128;

#pragma unroll
        for (int st = 0; st < 8; ++st) {
            const int mrow16 = st * 16 + (lane & 15);
            // Preload all four A chunks so the ds_loads pipeline and the
            // four WMMAs issue back-to-back on the accumulation chain.
            v16h a[4];
#pragma unroll
            for (int kc = 0; kc < 4; ++kc) {
                // k order used here (contiguous, as in B): lanes 0-15 hold K{0..15},
                // lanes 16-31 hold K{16..31}.
                const _Float16* lp =
                    lbase + mrow16 * 136 + kc * 32 + ((lane < 16) ? 0 : 16);
                v8h lo = *(const v8h*)(lp);
                v8h hi = *(const v8h*)(lp + 8);
                a[kc] = __builtin_shufflevector(lo, hi,
                    0,1,2,3,4,5,6,7,8,9,10,11,12,13,14,15);
            }
            v8f c = {};
#pragma unroll
            for (int kc = 0; kc < 4; ++kc)
                c = __builtin_amdgcn_wmma_f32_16x16x32_f16(
                        false, a[kc], false, breg[kc], (short)0, c, false, false);

            // 8 packed mask bytes for (anchor, m0+st*16 + mqoff*8 .. +7)
            const unsigned long long mv = mrow[(m0 + st * 16) / 8 + mqoff];
            npos += (int)__popcll(mv & 0x0101010101010101ULL);

            float v[8];
#pragma unroll
            for (int j = 0; j < 8; ++j) v[j] = c[j] * invT;

            float lmax = v[0];
#pragma unroll
            for (int j = 1; j < 8; ++j) lmax = fmaxf(lmax, v[j]);
            float nm  = fmaxf(mden, lmax);
            float add = 0.f;
#pragma unroll
            for (int j = 0; j < 8; ++j) add += __expf(v[j] - nm);
            sden = sden * __expf(mden - nm) + add;
            mden = nm;

            float lmaxn = NEG_INF;
#pragma unroll
            for (int j = 0; j < 8; ++j)
                if ((mv >> (8 * j)) & 1) lmaxn = fmaxf(lmaxn, v[j]);
            if (lmaxn != NEG_INF) {
                float nmn  = fmaxf(mnum, lmaxn);
                float addn = 0.f;
#pragma unroll
                for (int j = 0; j < 8; ++j)
                    addn += ((mv >> (8 * j)) & 1) ? __expf(v[j] - nmn) : 0.f;
                snum = snum * __expf(mnum - nmn) + addn;
                mnum = nmn;
            }
        }
        __syncthreads();   // all waves done with buffer t before it is restaged
    }

    // Merge the two half-rows of each anchor (lane a <-> lane a+16).
    const float mden2 = __shfl_xor(mden, 16, 32);
    const float sden2 = __shfl_xor(sden, 16, 32);
    const float mnum2 = __shfl_xor(mnum, 16, 32);
    const float snum2 = __shfl_xor(snum, 16, 32);
    const int   nptot = npos + __shfl_xor(npos, 16, 32);

    const float mmD = fmaxf(mden, mden2);
    const float den = logf(sden * __expf(mden - mmD) + sden2 * __expf(mden2 - mmD)) + mmD;
    const float mmN = fmaxf(mnum, mnum2);
    const float num = logf(snum * __expf(mnum - mmN) + snum2 * __expf(mnum2 - mmN)) + mmN;

    if (lane < 16 && nptot > 0 && nptot < M_DIM) {
        atomicAdd(&stemSum[s], den - num);
        atomicAdd(&stemCnt[s], 1);
    }
}

// ---------------------------------------------------------------------------
// Kernel 3: final reduction over stems.
// ---------------------------------------------------------------------------
__global__ void finalize_kernel(const float* __restrict__ stemSum,
                                const int* __restrict__ stemCnt,
                                float* __restrict__ out)
{
    if (threadIdx.x == 0 && blockIdx.x == 0) {
        float tot = 0.f; int sv = 0;
        for (int s = 0; s < S_DIM; ++s) {
            int c = stemCnt[s];
            if (c > 0) { tot += stemSum[s] / (float)c; ++sv; }
        }
        out[0] = (sv > 0) ? (tot / (float)sv) : 0.f;
    }
}

extern "C" void kernel_launch(void* const* d_in, const int* in_sizes, int n_in,
                              void* d_out, int out_size, void* d_ws, size_t ws_size,
                              hipStream_t stream)
{
    const float*         art  = (const float*)d_in[0];
    const float*         ref  = (const float*)d_in[1];
    const unsigned char* pm   = (const unsigned char*)d_in[2];  // jnp.bool_: 1 byte/elt
    const float*         logT = (const float*)d_in[3];
    float*               out  = (float*)d_out;

    char*     ws      = (char*)ws_size ? (char*)d_ws : (char*)d_ws;
    ws = (char*)d_ws;
    float*    stemSum = (float*)ws;                 // 16 B
    int*      stemCnt = (int*)(ws + 16);            // 16 B
    _Float16* aH      = (_Float16*)(ws + 64);                       // 4 MB
    _Float16* rH      = aH + (long long)S_DIM * N_DIM * D_DIM;      // 4 MB

    (void)hipMemsetAsync(ws, 0, 64, stream);

    {
        long long rows = (long long)S_DIM * (N_DIM + M_DIM);
        int blocks = (int)((rows + 7) / 8);
        nrm_kernel<<<blocks, 256, 0, stream>>>(art, ref, aH, rH);
    }
    {
        dim3 grid(N_DIM / 128, S_DIM);
        infonce_kernel<<<grid, 256, 0, stream>>>(aH, rH, pm, logT, stemSum, stemCnt);
    }
    finalize_kernel<<<1, 1, 0, stream>>>(stemSum, stemCnt, out);
}
